// GPT2Attention_66314295050289
// MI455X (gfx1250) — matrix operand order repaired and checked by reading, not compiled, not run on hardware
//
#include <hip/hip_runtime.h>

// ---------------------------------------------------------------------------
// GPT-2 attention for MI455X (gfx1250): bf16 WMMA GEMMs + flash attention,
// TDM (tensor_load_to_lds) staging of Q/K tiles where available.
// B=4, T=2048, C=1024, H=16, D=64.
// ---------------------------------------------------------------------------

typedef __attribute__((ext_vector_type(16))) __bf16 v16bf;
typedef __attribute__((ext_vector_type(8)))  float  v8f;

union Frag { v16bf v; unsigned u[8]; };
union Acc  { v8f  v; float    f[8]; };

__device__ __forceinline__ unsigned short f2bf(float f) {
  unsigned u = __float_as_uint(f);
  u += 0x7FFFu + ((u >> 16) & 1u);          // round-to-nearest-even
  return (unsigned short)(u >> 16);
}

__device__ __forceinline__ v8f wmma_bf16(const Frag& a, const Frag& b, v8f c) {
  return __builtin_amdgcn_wmma_f32_16x16x32_bf16(false, a.v, false, b.v,
                                                 (short)0, c, false, false);
}

// ---------------------------------------------------------------------------
// Tensor Data Mover: 2-D bf16 tile (rows x cols, row-major, contiguous into
// LDS).  D# per CDNA5 ISA §8.3/8.4: group0 {count=1, lds_addr, global_addr,
// type=2}, group1 {data_size=2B, tensor dims/strides, tile dims}.
// This toolchain exposes the 6-arg builtin:
//   (uint32x4 g0, int32x8 g1, int32x4, int32x4, int32x8, i32 cpol)
// ---------------------------------------------------------------------------
#if __has_builtin(__builtin_amdgcn_tensor_load_to_lds)
#define HAVE_TDM 1
typedef unsigned tdm_v4u __attribute__((ext_vector_type(4)));
typedef int      tdm_v8i __attribute__((ext_vector_type(8)));
typedef int      tdm_v4i __attribute__((ext_vector_type(4)));

__device__ __forceinline__ void tdm_load_2d_bf16(const void* gaddr,
                                                 unsigned lds_byte_off,
                                                 unsigned tensor_d0,
                                                 unsigned tensor_d1,
                                                 unsigned stride0_elems,
                                                 unsigned tile_d0,
                                                 unsigned tile_d1) {
  unsigned long long ga = (unsigned long long)gaddr;
  tdm_v4u g0;
  g0[0] = 1u;                                            // count=1, user mode
  g0[1] = lds_byte_off;                                  // lds_addr
  g0[2] = (unsigned)(ga & 0xFFFFFFFFull);                // global_addr[31:0]
  g0[3] = (unsigned)((ga >> 32) & 0x01FFFFFFull) | (2u << 30);  // [56:32]|type=2
  tdm_v8i g1;
  g1[0] = (int)(1u << 16);                               // data_size=1 (2 bytes)
  g1[1] = (int)((tensor_d0 & 0xFFFFu) << 16);            // tensor_dim0[15:0]
  g1[2] = (int)(((tensor_d0 >> 16) & 0xFFFFu) |
                ((tensor_d1 & 0xFFFFu) << 16));          // dim0[31:16]|dim1[15:0]
  g1[3] = (int)(((tensor_d1 >> 16) & 0xFFFFu) |
                ((tile_d0 & 0xFFFFu) << 16));            // dim1[31:16]|tile0
  g1[4] = (int)(tile_d1 & 0xFFFFu);                      // tile_dim1, tile_dim2=0
  g1[5] = (int)stride0_elems;                            // tensor_dim0_stride
  g1[6] = 0;
  g1[7] = 0;
  tdm_v4i gz4 = {0, 0, 0, 0};                            // 2-D: groups 2/3 unused
  tdm_v8i gz8 = {0, 0, 0, 0, 0, 0, 0, 0};
  __builtin_amdgcn_tensor_load_to_lds(g0, g1, gz4, gz4, gz8, 0);
}
#else
#define HAVE_TDM 0
#endif

// ---------------------------------------------------------------------------
// Tiled GEMM: C[M,N] = A[M,K] * Bw[K,N] + bias[N]
//   TA = float (convert to bf16 while staging) or unsigned short (raw bf16)
//   EPI = 0: scatter bf16 into Q/K/V [B*H, T, 64]   (QKV projection, N=3072)
//   EPI = 1: store f32 to outF                       (output projection)
// Block tile 128x128x32, 256 threads = 8 waves, wave tile 64x32 (4x2 WMMA).
// LDS: As bf16 [128][32] row-major; Bs bf16 [128][32] as [n][k] (transposed)
// so every fragment's 8 dwords form two contiguous 4-dword runs -> ds_load_b128.
// ---------------------------------------------------------------------------
__device__ __forceinline__ void stage_A_tile(const float* __restrict__ A,
                                             unsigned short* As, int blockM,
                                             int k0, int Kdim, int t) {
#pragma unroll
  for (int i = 0; i < 4; ++i) {
    int chunk = i * 256 + t;                 // 1024 chunks of 4 floats
    int row = chunk >> 3;                    // 8 chunks per row
    int col = (chunk & 7) * 4;
    const float4 v =
        *(const float4*)&A[(size_t)(blockM + row) * Kdim + k0 + col];
    uint2 pk;
    pk.x = (unsigned)f2bf(v.x) | ((unsigned)f2bf(v.y) << 16);
    pk.y = (unsigned)f2bf(v.z) | ((unsigned)f2bf(v.w) << 16);
    *(uint2*)&As[row * 32 + col] = pk;
  }
}
__device__ __forceinline__ void stage_A_tile(const unsigned short* __restrict__ A,
                                             unsigned short* As, int blockM,
                                             int k0, int Kdim, int t) {
#pragma unroll
  for (int i = 0; i < 2; ++i) {
    int chunk = i * 256 + t;                 // 512 chunks of 8 halfs
    int row = chunk >> 2;                    // 4 chunks per row
    int col = (chunk & 3) * 8;
    uint4 v = *(const uint4*)&A[(size_t)(blockM + row) * Kdim + k0 + col];
    *(uint4*)&As[row * 32 + col] = v;
  }
}

template<typename TA, int EPI>
__global__ __launch_bounds__(256)
void gemm_wmma_kernel(const TA* __restrict__ A, const float* __restrict__ Bw,
                      const float* __restrict__ bias,
                      unsigned short* __restrict__ outQ,
                      unsigned short* __restrict__ outK,
                      unsigned short* __restrict__ outV,
                      float* __restrict__ outF,
                      int Mdim, int Ndim, int Kdim)
{
  constexpr int BM = 128, BN = 128, BK = 32;
  __shared__ unsigned short As[BM * BK];     // [row][k]  bf16
  __shared__ unsigned short Bs16[BN * BK];   // [n][k]    bf16 (transposed)

  const int t    = threadIdx.x;
  const int lane = t & 31, w = t >> 5;
  const int wr = w >> 2, wc = w & 3;         // wave grid 2x4
  const int ln = lane & 15, lh = lane >> 4;
  const int blockN = blockIdx.x * BN, blockM = blockIdx.y * BM;

  Acc acc[4][2];
#pragma unroll
  for (int mt = 0; mt < 4; ++mt)
#pragma unroll
    for (int nt = 0; nt < 2; ++nt)
#pragma unroll
      for (int r = 0; r < 8; ++r) acc[mt][nt].f[r] = 0.0f;

  const unsigned* Asd = (const unsigned*)As;
  const unsigned* Bsd = (const unsigned*)Bs16;

  for (int k0 = 0; k0 < Kdim; k0 += BK) {
    __syncthreads();
    stage_A_tile(A, As, blockM, k0, Kdim, t);
    // ---- stage B tile transposed: Bs16[n*32 + k] ----
#pragma unroll
    for (int i = 0; i < 4; ++i) {
      int chunk = i * 256 + t;               // 1024 chunks of 4 floats
      int kk = chunk >> 5;                   // 32 chunks per k-row
      int n0 = (chunk & 31) * 4;
      const float4 v =
          *(const float4*)&Bw[(size_t)(k0 + kk) * Ndim + blockN + n0];
      Bs16[(n0 + 0) * 32 + kk] = f2bf(v.x);
      Bs16[(n0 + 1) * 32 + kk] = f2bf(v.y);
      Bs16[(n0 + 2) * 32 + kk] = f2bf(v.z);
      Bs16[(n0 + 3) * 32 + kk] = f2bf(v.w);
    }
    if (k0 + BK < Kdim) {  // speculative prefetch of next tiles
      __builtin_prefetch(&A[(size_t)(blockM + (t >> 1)) * Kdim + k0 + BK], 0, 0);
      __builtin_prefetch(&Bw[(size_t)(k0 + BK + (t >> 3)) * Ndim + blockN], 0, 0);
    }
    __syncthreads();

    // ---- fragments (two contiguous 4-dword runs per fragment -> b128 pairs) ----
    Frag af[4];
#pragma unroll
    for (int mt = 0; mt < 4; ++mt)
#pragma unroll
      for (int d = 0; d < 8; ++d) {
        int dw = d + 4 * lh + ((d & 4) ? 4 : 0);   // A: K=2d+8*lh+8*(d>=4)
        af[mt].u[d] = Asd[(wr * 64 + mt * 16 + ln) * 16 + dw];
      }
    Frag bfr[2];
#pragma unroll
    for (int nt = 0; nt < 2; ++nt) {
      int n = wc * 32 + nt * 16 + ln;
#pragma unroll
      for (int d = 0; d < 8; ++d)                  // B: K=2d+8*lh+8*(d>=4)
        bfr[nt].u[d] = Bsd[n * 16 + d + 4 * lh + ((d & 4) ? 4 : 0)];
    }
#pragma unroll
    for (int mt = 0; mt < 4; ++mt)
#pragma unroll
      for (int nt = 0; nt < 2; ++nt)
        acc[mt][nt].v = wmma_bf16(af[mt], bfr[nt], acc[mt][nt].v);
  }

  // ---- epilogue ----
#pragma unroll
  for (int mt = 0; mt < 4; ++mt)
#pragma unroll
    for (int nt = 0; nt < 2; ++nt)
#pragma unroll
      for (int r = 0; r < 8; ++r) {
        int row = blockM + wr * 64 + mt * 16 + r + 8 * lh;  // C: M=r+8*lh
        int col = blockN + wc * 32 + nt * 16 + ln;          //    N=lane%16
        float val = acc[mt][nt].f[r] + bias[col];
        if (EPI == 0) {
          int which = col >> 10;            // 0=q 1=k 2=v
          int c     = col & 1023;
          int h     = c >> 6;
          int dcol  = c & 63;
          int b     = row >> 11;            // T = 2048
          int tt    = row & 2047;
          size_t dst = ((size_t)(b * 16 + h) * 2048 + tt) * 64 + dcol;
          unsigned short* o = (which == 0) ? outQ : (which == 1) ? outK : outV;
          o[dst] = f2bf(val);
        } else {
          outF[(size_t)row * Ndim + col] = val;
        }
      }
}

// ---------------------------------------------------------------------------
// Flash attention: one workgroup = 128 query rows of one (b,h) head.
// 8 waves; each wave owns 16 query rows with private online softmax state.
// Q block + K tiles staged by TDM (wave 0) while other waves stage V.
// Writes y as bf16 [B*T, C].
// ---------------------------------------------------------------------------
__global__ __launch_bounds__(256)
void flash_attn_kernel(const unsigned short* __restrict__ Q,
                       const unsigned short* __restrict__ K,
                       const unsigned short* __restrict__ V,
                       unsigned short* __restrict__ Y)
{
  constexpr int T = 2048, D = 64, Cc = 1024;
  __shared__ unsigned short Qs[128 * 64];    // [qrow][d]   row-major
  __shared__ unsigned short Ks[64 * 64];     // [kvrow][d]  row-major
  __shared__ unsigned short Vs16[64 * 64];   // [d][kv]     transposed
  __shared__ unsigned short Ps[8][16 * 64];  // per-wave P [qrow][kv]

  const int t    = threadIdx.x;
  const int lane = t & 31, w = t >> 5;
  const int ln = lane & 15, lh = lane >> 4;
  const int bh    = blockIdx.y;
  const int qbase = blockIdx.x * 128;
  const size_t headoff = (size_t)bh * T * D;

  // ---- stage Q block ----
#if HAVE_TDM
  if (w == 0) {
    tdm_load_2d_bf16(Q + headoff + (size_t)qbase * D,
                     (unsigned)(size_t)(void*)Qs, D, T, D, D, 128);
    __builtin_amdgcn_s_wait_tensorcnt(0);
  }
#else
#pragma unroll
  for (int i = 0; i < 4; ++i) {
    int chunk = i * 256 + t;                 // 1024 chunks of 8 halfs
    int row = chunk >> 3, col = (chunk & 7) * 8;
    *(uint4*)&Qs[row * 64 + col] =
        *(const uint4*)&Q[headoff + (size_t)(qbase + row) * D + col];
  }
#endif
  __syncthreads();

  // ---- per-wave persistent Q fragments ----
  Frag qf[2];
  const unsigned* Qsd = (const unsigned*)Qs;
  const int qm = w * 16 + ln;
#pragma unroll
  for (int c = 0; c < 2; ++c)
#pragma unroll
    for (int d = 0; d < 8; ++d) {
      int dw = d + 4 * lh + ((d & 4) ? 4 : 0) + c * 16;
      qf[c].u[d] = Qsd[qm * 32 + dw];
    }

  float mi[8], li[8];
  Acc oacc[4];
#pragma unroll
  for (int r = 0; r < 8; ++r) { mi[r] = -1e30f; li[r] = 0.0f; }
#pragma unroll
  for (int nt = 0; nt < 4; ++nt)
#pragma unroll
    for (int r = 0; r < 8; ++r) oacc[nt].f[r] = 0.0f;

  const float scale = 0.125f;                // 1/sqrt(64)
  const unsigned* Psd = (const unsigned*)Ps[w];
  const unsigned* Ksd = (const unsigned*)Ks;
  const unsigned* Vsd = (const unsigned*)Vs16;

  for (int kv = 0; kv < T; kv += 64) {
    __syncthreads();
    // ---- K tile via TDM (wave 0) overlapped with V staging (all waves) ----
#if HAVE_TDM
    if (w == 0)
      tdm_load_2d_bf16(K + headoff + (size_t)kv * D,
                       (unsigned)(size_t)(void*)Ks, D, T, D, D, 64);
#else
#pragma unroll
    for (int i = 0; i < 2; ++i) {
      int chunk = i * 256 + t;               // 512 chunks of 8 halfs
      int row = chunk >> 3, col = (chunk & 7) * 8;
      *(uint4*)&Ks[row * 64 + col] =
          *(const uint4*)&K[headoff + (size_t)(kv + row) * D + col];
    }
#endif
    // ---- V tile transposed: Vs16[n*64 + kk] ----
#pragma unroll
    for (int i = 0; i < 4; ++i) {
      int chunk = i * 256 + t;               // 1024 chunks of 4 halfs
      int kk = chunk >> 4;                   // 16 chunks per kv row
      int n0 = (chunk & 15) * 4;
      uint2 v = *(const uint2*)&V[headoff + (size_t)(kv + kk) * D + n0];
      Vs16[(n0 + 0) * 64 + kk] = (unsigned short)(v.x & 0xFFFFu);
      Vs16[(n0 + 1) * 64 + kk] = (unsigned short)(v.x >> 16);
      Vs16[(n0 + 2) * 64 + kk] = (unsigned short)(v.y & 0xFFFFu);
      Vs16[(n0 + 3) * 64 + kk] = (unsigned short)(v.y >> 16);
    }
#if HAVE_TDM
    if (w == 0) __builtin_amdgcn_s_wait_tensorcnt(0);
#endif
    __syncthreads();

    // ---- S = Q * K^T  (16 x 64 per wave) ----
    Acc sacc[4];
#pragma unroll
    for (int nt = 0; nt < 4; ++nt) {
#pragma unroll
      for (int r = 0; r < 8; ++r) sacc[nt].f[r] = 0.0f;
#pragma unroll
      for (int c = 0; c < 2; ++c) {
        Frag bfr;
#pragma unroll
        for (int d = 0; d < 8; ++d)
          bfr.u[d] = Ksd[(nt * 16 + ln) * 32 + d + 4 * lh + ((d & 4) ? 4 : 0) + c * 16];
        sacc[nt].v = wmma_bf16(qf[c], bfr, sacc[nt].v);
      }
#pragma unroll
      for (int r = 0; r < 8; ++r) sacc[nt].f[r] *= scale;
    }

    // ---- online softmax (rows r, r+8 live in the two 16-lane halves) ----
    float mnew[8], alpha[8];
#pragma unroll
    for (int r = 0; r < 8; ++r) {
      float mx = fmaxf(fmaxf(sacc[0].f[r], sacc[1].f[r]),
                       fmaxf(sacc[2].f[r], sacc[3].f[r]));
#pragma unroll
      for (int m = 1; m < 16; m <<= 1) mx = fmaxf(mx, __shfl_xor(mx, m, 32));
      mnew[r]  = fmaxf(mi[r], mx);
      alpha[r] = __expf(mi[r] - mnew[r]);
      mi[r]    = mnew[r];
    }
    float rs[8];
#pragma unroll
    for (int r = 0; r < 8; ++r) rs[r] = 0.0f;
#pragma unroll
    for (int nt = 0; nt < 4; ++nt)
#pragma unroll
      for (int r = 0; r < 8; ++r) {
        float p = __expf(sacc[nt].f[r] - mnew[r]);
        rs[r] += p;
        Ps[w][(r + 8 * lh) * 64 + nt * 16 + ln] = f2bf(p);
      }
#pragma unroll
    for (int r = 0; r < 8; ++r) {
      float s = rs[r];
#pragma unroll
      for (int m = 1; m < 16; m <<= 1) s += __shfl_xor(s, m, 32);
      li[r] = li[r] * alpha[r] + s;
#pragma unroll
      for (int nt = 0; nt < 4; ++nt) oacc[nt].f[r] *= alpha[r];
    }
    __syncthreads();   // Ps visible across lanes before A-fragment reads

    // ---- O += P * V ----
#pragma unroll
    for (int c = 0; c < 2; ++c) {
      Frag pf;
#pragma unroll
      for (int d = 0; d < 8; ++d) {
        int dw = d + 4 * lh + ((d & 4) ? 4 : 0) + c * 16;
        pf.u[d] = Psd[ln * 32 + dw];
      }
#pragma unroll
      for (int nt = 0; nt < 4; ++nt) {
        Frag vf;
        int n = nt * 16 + ln;
#pragma unroll
        for (int d = 0; d < 8; ++d)
          vf.u[d] = Vsd[n * 32 + c * 16 + d + 4 * lh + ((d & 4) ? 4 : 0)];
        oacc[nt].v = wmma_bf16(pf, vf, oacc[nt].v);
      }
    }
  }

  // ---- normalize and write y (bf16, [B*T, C]) ----
  const int b = bh >> 4, h = bh & 15;
#pragma unroll
  for (int nt = 0; nt < 4; ++nt)
#pragma unroll
    for (int r = 0; r < 8; ++r) {
      int qrow = qbase + w * 16 + r + 8 * lh;
      int col  = h * 64 + nt * 16 + ln;
      float val = oacc[nt].f[r] / li[r];
      Y[(size_t)(b * T + qrow) * Cc + col] = f2bf(val);
    }
}

// ---------------------------------------------------------------------------
extern "C" void kernel_launch(void* const* d_in, const int* in_sizes, int n_in,
                              void* d_out, int out_size, void* d_ws, size_t ws_size,
                              hipStream_t stream) {
  const float* x      = (const float*)d_in[0];
  const float* W_attn = (const float*)d_in[1];
  const float* b_attn = (const float*)d_in[2];
  const float* W_proj = (const float*)d_in[3];
  const float* b_proj = (const float*)d_in[4];
  float* out = (float*)d_out;

  const size_t NEL = (size_t)8192 * 1024;   // B*T*C elements
  unsigned short* Q = (unsigned short*)d_ws;
  unsigned short* K = Q + NEL;
  unsigned short* V = K + NEL;
  unsigned short* Y = V + NEL;

  // 1) QKV projection: [8192,1024] x [1024,3072] -> scatter bf16 Q/K/V
  dim3 g1(3072 / 128, 8192 / 128);
  gemm_wmma_kernel<float, 0><<<g1, 256, 0, stream>>>(
      x, W_attn, b_attn, Q, K, V, nullptr, 8192, 3072, 1024);

  // 2) flash attention over 64 heads, 16 query blocks each
  dim3 g2(2048 / 128, 4 * 16);
  flash_attn_kernel<<<g2, 256, 0, stream>>>(Q, K, V, Y);

  // 3) output projection: [8192,1024] x [1024,1024] -> f32 d_out
  dim3 g3(1024 / 128, 8192 / 128);
  gemm_wmma_kernel<unsigned short, 1><<<g3, 256, 0, stream>>>(
      Y, W_proj, b_proj, nullptr, nullptr, nullptr, out, 8192, 1024, 1024);
}
